// VectorQuantizer_49074296324716
// MI455X (gfx1250) — compile-verified, matrix operand order checked by reading, not run on hardware
//
#include <hip/hip_runtime.h>
#include <hip/hip_bf16.h>

// ---------------------------------------------------------------------------
// VectorQuantizer (Sinkhorn VQ) for MI455X / gfx1250, wave32 + WMMA.
//
// cos = l2norm(x) @ l2norm(cb)^T via split-precision f16 WMMA:
//   x = hi + lo (both f16);  cos ~= Ahi*Bhi + Alo*Bhi + Ahi*Blo  (err ~2e-7,
//   i.e. f32-grade, while staying on v_wmma_f32_16x16x32_f16).
// Sinkhorn runs in the log domain (f32-safe; the reference needs f64 only
// because it exponentiates arguments up to +-200 directly).
// indices = argmax_k (alpha*cos[n,k] + u_k) after 3 row updates.
// loss = mean(LSE_k(cos/tau) - cos[n,idx]/tau).  x_q = cb[idx].
// cos[N,K] f32 = 134 MB -> resident in the 192 MB L2 across all 6 passes.
// ---------------------------------------------------------------------------

#define NPTS 32768
#define DIM  128
#define KCB  1024
#define NSPL 64          // N-chunks for row-LSE partials (chunk = 512 rows)

static __device__ __constant__ float kTau    = 0.2f;
static __device__ __constant__ float kSkEps  = 0.005f;

typedef __attribute__((ext_vector_type(16))) _Float16 v16h;
typedef __attribute__((ext_vector_type(8)))  _Float16 h8v;
typedef __attribute__((ext_vector_type(4)))  _Float16 h4v;
typedef __attribute__((ext_vector_type(8)))  float    v8f;

struct Scal {
    unsigned minEnc;   // ordered-encoded min of cos
    unsigned maxEnc;   // ordered-encoded max of cos
    float    alpha;    // f = alpha*cos + beta  (== -dc/eps)
    float    beta;
    float    loss;     // sum of per-row CE losses
};

// ---- ordered float<->uint encoding (monotonic) ----------------------------
__device__ inline unsigned f2ord(float f) {
    unsigned u = __float_as_uint(f);
    return (u & 0x80000000u) ? ~u : (u | 0x80000000u);
}
__device__ inline float ord2f(unsigned e) {
    unsigned b = (e & 0x80000000u) ? (e ^ 0x80000000u) : ~e;
    return __uint_as_float(b);
}

__device__ inline v16h mk16(h8v lo, h8v hi) {
    v16h r;
#pragma unroll
    for (int i = 0; i < 8; ++i) { r[i] = lo[i]; r[i + 8] = hi[i]; }
    return r;
}

// ---------------------------------------------------------------------------
__global__ __launch_bounds__(64) void init_kernel(Scal* sc) {
    if (threadIdx.x == 0) {
        sc->minEnc = 0xFFFFFFFFu;   // encoded +inf
        sc->maxEnc = 0x00000000u;   // encoded -inf
        sc->loss   = 0.0f;
    }
}

// ---- wave-per-row L2 normalize + f32 -> (hi,lo) f16 pair ------------------
__global__ __launch_bounds__(256) void normalize_kernel(const float* __restrict__ src,
                                                        _Float16* __restrict__ dhi,
                                                        _Float16* __restrict__ dlo,
                                                        int rows) {
    int wid  = (blockIdx.x * blockDim.x + threadIdx.x) >> 5;
    int lane = threadIdx.x & 31;
    if (wid >= rows) return;
    const float4* s4 = reinterpret_cast<const float4*>(src + (size_t)wid * DIM);
    float4 x = s4[lane];                         // 32 lanes * 4 = 128
    float ss = x.x * x.x + x.y * x.y + x.z * x.z + x.w * x.w;
#pragma unroll
    for (int off = 16; off > 0; off >>= 1) ss += __shfl_xor(ss, off, 32);
    float scale = 1.0f / fmaxf(sqrtf(ss), 1e-12f);
    float v[4] = {x.x * scale, x.y * scale, x.z * scale, x.w * scale};
    h4v hi, lo;
#pragma unroll
    for (int i = 0; i < 4; ++i) {
        hi[i] = (_Float16)v[i];
        lo[i] = (_Float16)(v[i] - (float)hi[i]);
    }
    reinterpret_cast<h4v*>(dhi)[(size_t)wid * 32 + lane] = hi;
    reinterpret_cast<h4v*>(dlo)[(size_t)wid * 32 + lane] = lo;
}

// ---- WMMA GEMM: cos[N,K] = Xn @ Cn^T (split f16), plus min/max(cos) -------
// grid (KCB/128, NPTS/128), block 256 (8 waves). Codebook tile in LDS,
// staged twice (hi then lo) through the same 34KB buffer.
#define BPITCH 136   // halfs per LDS row (128 + 8 pad -> 272B, conflict-free)
__global__ __launch_bounds__(256) void gemm_kernel(const _Float16* __restrict__ Xhi,
                                                   const _Float16* __restrict__ Xlo,
                                                   const _Float16* __restrict__ Chi,
                                                   const _Float16* __restrict__ Clo,
                                                   float* __restrict__ cosm,
                                                   Scal* sc) {
    __shared__ _Float16 sB[128 * BPITCH];
    __shared__ float smin[8], smax[8];

    const int tid  = threadIdx.x;
    const int Nblk = blockIdx.x * 128;   // codebook-row block (output cols)
    const int Mblk = blockIdx.y * 128;   // x-row block

    const int wave = tid >> 5, lane = tid & 31;
    const int r = lane & 15, h = lane >> 4;
    const int m0 = Mblk + wave * 16;

    auto loadTile = [&](const _Float16* src) {
#pragma unroll
        for (int j = 0; j < 8; ++j) {
            int c = tid + j * 256;           // half8 chunk id, 2048 total
            int rr = c >> 4, d = (c & 15) * 8;
            *reinterpret_cast<h8v*>(&sB[rr * BPITCH + d]) =
                *reinterpret_cast<const h8v*>(src + (size_t)(Nblk + rr) * DIM + d);
        }
    };
    auto ldB = [&](int t, int kt) -> v16h {
        const h8v* Brow = reinterpret_cast<const h8v*>(&sB[(t * 16 + r) * BPITCH]);
        return mk16(Brow[kt * 4 + h], Brow[kt * 4 + 2 + h]);
    };

    // A fragments: 16 rows x full K=128 for hi and lo, kept in registers
    const h8v* Ah = reinterpret_cast<const h8v*>(Xhi + (size_t)(m0 + r) * DIM);
    const h8v* Al = reinterpret_cast<const h8v*>(Xlo + (size_t)(m0 + r) * DIM);
    v16h ahi[4], alo[4];
#pragma unroll
    for (int kt = 0; kt < 4; ++kt) {
        ahi[kt] = mk16(Ah[kt * 4 + h], Ah[kt * 4 + 2 + h]);
        alo[kt] = mk16(Al[kt * 4 + h], Al[kt * 4 + 2 + h]);
    }

    v8f acc[8];
#pragma unroll
    for (int t = 0; t < 8; ++t) acc[t] = (v8f){};

    // ---- stage 1: B = Chi tile;  acc += Ahi*Bhi + Alo*Bhi ----
    loadTile(Chi);
    __syncthreads();
    {
        v16h bnext = ldB(0, 0);
#pragma unroll
        for (int t = 0; t < 8; ++t) {
#pragma unroll
            for (int kt = 0; kt < 4; ++kt) {
                v16h b = bnext;
                int nx = t * 4 + kt + 1;
                if (nx < 32) bnext = ldB(nx >> 2, nx & 3);   // prefetch next frag
                acc[t] = __builtin_amdgcn_wmma_f32_16x16x32_f16(
                    false, alo[kt], false, b, (short)0, acc[t], false, false);
                acc[t] = __builtin_amdgcn_wmma_f32_16x16x32_f16(
                    false, ahi[kt], false, b, (short)0, acc[t], false, false);
            }
        }
    }
    __syncthreads();

    // ---- stage 2: B = Clo tile;  acc += Ahi*Blo ----
    loadTile(Clo);
    __syncthreads();
    {
        v16h bnext = ldB(0, 0);
#pragma unroll
        for (int t = 0; t < 8; ++t) {
#pragma unroll
            for (int kt = 0; kt < 4; ++kt) {
                v16h b = bnext;
                int nx = t * 4 + kt + 1;
                if (nx < 32) bnext = ldB(nx >> 2, nx & 3);
                acc[t] = __builtin_amdgcn_wmma_f32_16x16x32_f16(
                    false, ahi[kt], false, b, (short)0, acc[t], false, false);
            }
        }
    }

    // ---- epilogue: store cos tiles + block min/max ----
    float lmin = 3.4e38f, lmax = -3.4e38f;
#pragma unroll
    for (int t = 0; t < 8; ++t) {
        const int col = Nblk + t * 16 + r;
#pragma unroll
        for (int v = 0; v < 8; ++v) {
            float val = acc[t][v];
            cosm[(size_t)(m0 + v + 8 * h) * KCB + col] = val;
            lmin = fminf(lmin, val); lmax = fmaxf(lmax, val);
        }
    }
#pragma unroll
    for (int off = 16; off > 0; off >>= 1) {
        lmin = fminf(lmin, __shfl_xor(lmin, off, 32));
        lmax = fmaxf(lmax, __shfl_xor(lmax, off, 32));
    }
    if (lane == 0) { smin[wave] = lmin; smax[wave] = lmax; }
    __syncthreads();
    if (tid == 0) {
        float bmin = smin[0], bmax = smax[0];
#pragma unroll
        for (int w = 1; w < 8; ++w) { bmin = fminf(bmin, smin[w]); bmax = fmaxf(bmax, smax[w]); }
        atomicMin(&sc->minEnc, f2ord(bmin));
        atomicMax(&sc->maxEnc, f2ord(bmax));
    }
}

// ---- centering constants: f = alpha*cos + beta == -dc/eps -----------------
__global__ void compute_ab_kernel(Scal* sc) {
    float cmin = ord2f(sc->minEnc), cmax = ord2f(sc->maxEnc);
    float dmax = (1.0f - cmin) / kTau, dmin = (1.0f - cmax) / kTau;
    float middle = 0.5f * (dmax + dmin);
    float amp = fmaxf(dmax - middle + 1e-5f, 1e-5f);
    sc->alpha = 1.0f / (kTau * amp * kSkEps);
    sc->beta  = (middle - 1.0f / kTau) / (amp * kSkEps);
}

__device__ inline void lse_push(float& m, float& s, float val) {
    float nm = fmaxf(m, val);
    s = s * __expf(m - nm) + __expf(val - nm);
    m = nm;
}

// ---- row-LSE partials: per k, LSE over a 512-row chunk of (f + v_n) -------
// grid (KCB/256, NSPL), block 256
__global__ __launch_bounds__(256) void row_lse_partial_kernel(const float* __restrict__ cosm,
                                                              const float* __restrict__ v,
                                                              int useV, const Scal* sc,
                                                              float* __restrict__ pmax,
                                                              float* __restrict__ psum) {
    const int k = blockIdx.x * 256 + threadIdx.x;
    const int n0 = blockIdx.y * (NPTS / NSPL);
    const float alpha = sc->alpha, beta = sc->beta;
    float m = -3.4e38f, s = 0.0f;
    for (int n = n0; n < n0 + NPTS / NSPL; ++n) {
        float add = useV ? v[n] : 0.0f;
        lse_push(m, s, alpha * cosm[(size_t)n * KCB + k] + beta + add);
    }
    pmax[blockIdx.y * KCB + k] = m;
    psum[blockIdx.y * KCB + k] = s;
}

// ---- combine partials -> u_k = -LSE - ln(K) -------------------------------
__global__ __launch_bounds__(256) void row_lse_combine_kernel(const float* __restrict__ pmax,
                                                              const float* __restrict__ psum,
                                                              float* __restrict__ u) {
    const int k = blockIdx.x * 256 + threadIdx.x;
    float M = -3.4e38f, S = 0.0f;
    for (int j = 0; j < NSPL; ++j) {
        float mj = pmax[j * KCB + k], sj = psum[j * KCB + k];
        float nm = fmaxf(M, mj);
        S = S * __expf(M - nm) + sj * __expf(mj - nm);
        M = nm;
    }
    u[k] = -(M + __logf(S)) - 6.9314718056f;   // ln(1024)
}

// ---- col-LSE: v_n = -LSE_k(f + u_k) - ln(N); wave per row -----------------
__global__ __launch_bounds__(256) void col_lse_kernel(const float* __restrict__ cosm,
                                                      const float* __restrict__ u,
                                                      const Scal* sc,
                                                      float* __restrict__ v) {
    const int n = blockIdx.x * 8 + (threadIdx.x >> 5);
    const int lane = threadIdx.x & 31;
    const float alpha = sc->alpha, beta = sc->beta;
    float m = -3.4e38f, s = 0.0f;
    const float4* c4 = reinterpret_cast<const float4*>(cosm + (size_t)n * KCB);
    const float4* u4 = reinterpret_cast<const float4*>(u);
#pragma unroll
    for (int j = 0; j < 8; ++j) {
        int i = j * 32 + lane;
        float4 c = c4[i], uu = u4[i];
        lse_push(m, s, alpha * c.x + beta + uu.x);
        lse_push(m, s, alpha * c.y + beta + uu.y);
        lse_push(m, s, alpha * c.z + beta + uu.z);
        lse_push(m, s, alpha * c.w + beta + uu.w);
    }
#pragma unroll
    for (int off = 16; off > 0; off >>= 1) {
        float om = __shfl_xor(m, off, 32), os = __shfl_xor(s, off, 32);
        float nm = fmaxf(m, om);
        s = s * __expf(m - nm) + os * __expf(om - nm);
        m = nm;
    }
    if (lane == 0) v[n] = -(m + __logf(s)) - 10.3972077084f;   // ln(32768)
}

// ---- finalize: argmax, CE loss, x_q gather, indices -----------------------
__global__ __launch_bounds__(256) void finalize_kernel(const float* __restrict__ cosm,
                                                       const float* __restrict__ u,
                                                       const Scal* scs,
                                                       const float* __restrict__ codebook,
                                                       float* __restrict__ xq,
                                                       float* __restrict__ lossAcc,
                                                       float* __restrict__ outIdx) {
    const int n = blockIdx.x * 8 + (threadIdx.x >> 5);
    const int lane = threadIdx.x & 31;
    const float alpha = scs->alpha, beta = scs->beta, invTau = 1.0f / kTau;
    float bf = -3.4e38f, bc = 0.0f;
    int bk = KCB;
    float lm = -3.4e38f, ls = 0.0f;
    const float4* c4 = reinterpret_cast<const float4*>(cosm + (size_t)n * KCB);
    const float4* u4 = reinterpret_cast<const float4*>(u);
#pragma unroll
    for (int j = 0; j < 8; ++j) {
        int i = j * 32 + lane;
        float4 c = c4[i], uu = u4[i];
        float cv[4] = {c.x, c.y, c.z, c.w};
        float uv[4] = {uu.x, uu.y, uu.z, uu.w};
#pragma unroll
        for (int e = 0; e < 4; ++e) {
            float f = alpha * cv[e] + beta + uv[e];
            int k = i * 4 + e;
            if (f > bf) { bf = f; bk = k; bc = cv[e]; }   // ks increasing: first max
            lse_push(lm, ls, cv[e] * invTau);
        }
    }
#pragma unroll
    for (int off = 16; off > 0; off >>= 1) {
        float of = __shfl_xor(bf, off, 32);
        int   ok = __shfl_xor(bk, off, 32);
        float oc = __shfl_xor(bc, off, 32);
        if (of > bf || (of == bf && ok < bk)) { bf = of; bk = ok; bc = oc; }
        float om = __shfl_xor(lm, off, 32), os = __shfl_xor(ls, off, 32);
        float nm = fmaxf(lm, om);
        ls = ls * __expf(lm - nm) + os * __expf(om - nm);
        lm = nm;
    }
    if (lane == 0) {
        atomicAdd(lossAcc, (lm + __logf(ls)) - bc * invTau);
        outIdx[n] = (float)bk;
    }
    // x_q row: gather original f32 codebook row bk (all lanes agree on bk)
    const float4* cb4 = reinterpret_cast<const float4*>(codebook + (size_t)bk * DIM);
    reinterpret_cast<float4*>(xq)[(size_t)n * 32 + lane] = cb4[lane];
}

__global__ void loss_div_kernel(const Scal* sc, float* outLoss) {
    *outLoss = sc->loss * (1.0f / (float)NPTS);
}

// ---------------------------------------------------------------------------
extern "C" void kernel_launch(void* const* d_in, const int* in_sizes, int n_in,
                              void* d_out, int out_size, void* d_ws, size_t ws_size,
                              hipStream_t stream) {
    const float* x  = (const float*)d_in[0];   // [N, D]
    const float* cb = (const float*)d_in[1];   // [K, D]
    float* out = (float*)d_out;                // [N*D] x_q | [1] loss | [N] indices

    // ---- workspace carve-up (all 256B aligned) ----
    char* base = (char*)d_ws;
    size_t off = 0;
    auto carve = [&](size_t bytes) { char* p = base + off; off = (off + bytes + 255) & ~(size_t)255; return p; };
    _Float16* Xhi = (_Float16*)carve((size_t)NPTS * DIM * 2);   // 8 MB
    _Float16* Xlo = (_Float16*)carve((size_t)NPTS * DIM * 2);   // 8 MB
    _Float16* Chi = (_Float16*)carve((size_t)KCB  * DIM * 2);   // 256 KB
    _Float16* Clo = (_Float16*)carve((size_t)KCB  * DIM * 2);   // 256 KB
    float*    cosm= (float*)   carve((size_t)NPTS * KCB * 4);   // 128 MB
    float*    u   = (float*)   carve((size_t)KCB * 4);
    float*    v   = (float*)   carve((size_t)NPTS * 4);
    float*    pmax= (float*)   carve((size_t)NSPL * KCB * 4);
    float*    psum= (float*)   carve((size_t)NSPL * KCB * 4);
    Scal*     sc  = (Scal*)    carve(sizeof(Scal));
    (void)ws_size; (void)in_sizes; (void)n_in; (void)out_size;

    float* xq      = out;
    float* outLoss = out + (size_t)NPTS * DIM;
    float* outIdx  = outLoss + 1;

    init_kernel<<<1, 64, 0, stream>>>(sc);
    normalize_kernel<<<NPTS / 8, 256, 0, stream>>>(x,  Xhi, Xlo, NPTS);
    normalize_kernel<<<KCB  / 8, 256, 0, stream>>>(cb, Chi, Clo, KCB);

    gemm_kernel<<<dim3(KCB / 128, NPTS / 128), 256, 0, stream>>>(Xhi, Xlo, Chi, Clo, cosm, sc);
    compute_ab_kernel<<<1, 1, 0, stream>>>(sc);

    // Sinkhorn, log domain. iter1 row (v=0) -> col ; iter2 row -> col ; iter3 row
    for (int it = 0; it < 3; ++it) {
        row_lse_partial_kernel<<<dim3(KCB / 256, NSPL), 256, 0, stream>>>(
            cosm, v, it > 0 ? 1 : 0, sc, pmax, psum);
        row_lse_combine_kernel<<<KCB / 256, 256, 0, stream>>>(pmax, psum, u);
        if (it < 2)
            col_lse_kernel<<<NPTS / 8, 256, 0, stream>>>(cosm, u, sc, v);
    }

    finalize_kernel<<<NPTS / 8, 256, 0, stream>>>(cosm, u, sc, cb, xq, &sc->loss, outIdx);
    loss_div_kernel<<<1, 1, 0, stream>>>(sc, outLoss);
}
